// TGNModel_47493748359504
// MI455X (gfx1250) — matrix operand order repaired and checked by reading, not compiled, not run on hardware
//
#include <hip/hip_runtime.h>
#include <hip/hip_bf16.h>

// ---------------------------------------------------------------------------
// TGN forward for MI455X (gfx1250, wave32). All dense contractions use
// v_wmma_f32_16x16x32_f16 (f16 operands, f32 accumulate). Weights are packed
// once into B-fragment-ready order so each lane's fragment is one contiguous
// 32-byte load. Workspace assumes ws_size >= ~210 MB.
// ---------------------------------------------------------------------------

typedef __attribute__((ext_vector_type(16))) _Float16 v16h;
typedef __attribute__((ext_vector_type(8)))  float    v8f;

// ---- WMMA fragment helpers (ISA 7.12.2 layouts, wave32) -------------------

// A: 16x32 f16 tile from an LDS row-major [16][KPITCH] buffer.
// lane L: M = L%16 ; halves 0..7 -> K = kt*32 + (L>=16?8:0) + j
//                    halves 8..15 -> K = kt*32 + 16 + (L>=16?8:0) + j
__device__ __forceinline__ v16h wmma_load_a(const _Float16* at, int lane, int kt, int kpitch) {
  const int m  = lane & 15;
  const int kb = kt * 32 + ((lane & 16) ? 8 : 0);
  const _Float16* p = at + m * kpitch + kb;
  v16h a;
#pragma unroll
  for (int j = 0; j < 8; ++j) a[j] = p[j];
#pragma unroll
  for (int j = 0; j < 8; ++j) a[j + 8] = p[16 + j];
  return a;
}

// B fragment from packed weight: one contiguous 32-byte load per lane.
// Packed layout: Wpk[((nt*nkt + kt)*32 + lane)*16 + j] = W[K][n] with
// K = kt*32 + (j&8?16:0) + (lane&16?8:0) + (j&7) , n = nt*16 + lane%16.
__device__ __forceinline__ v16h wmma_load_b_pk(const _Float16* Wpk, int lane,
                                               int kt, int nt, int nkt) {
  const _Float16* p = Wpk + (((size_t)(nt * nkt + kt)) * 32 + lane) * 16;
  return *(const v16h*)p;
}

__device__ __forceinline__ v8f wmma_mac(v16h a, v16h b, v8f acc) {
  return __builtin_amdgcn_wmma_f32_16x16x32_f16(false, a, false, b,
                                                (short)0, acc, false, false);
}

__device__ __forceinline__ void atomicMaxF(float* addr, float val) {
  if (val >= 0.f) atomicMax((int*)addr, __float_as_int(val));
  else            atomicMin((unsigned int*)addr, __float_as_uint(val));
}

// ---- weight packing: f32 [rows][cols] -> fragment-ordered f16 -------------
__global__ void tgn_pack_w_frag(const float* __restrict__ W, _Float16* __restrict__ Wpk,
                                int rows, int cols, int nkt, int total) {
  int i = blockIdx.x * blockDim.x + threadIdx.x;
  if (i >= total) return;
  int j    = i & 15;
  int lane = (i >> 4) & 31;
  int ktnt = i >> 9;           // nt*nkt + kt
  int kt   = ktnt % nkt;
  int nt   = ktnt / nkt;
  int K = kt * 32 + ((j & 8) ? 16 : 0) + ((lane & 16) ? 8 : 0) + (j & 7);
  int n = nt * 16 + (lane & 15);
  float v = (K < rows && n < cols) ? W[(size_t)K * cols + n] : 0.f;
  Wpk[i] = (_Float16)v;
}

__global__ void tgn_init(float* __restrict__ segmax, float* __restrict__ segsum,
                         float* __restrict__ outacc, int N) {
  int i = blockIdx.x * blockDim.x + threadIdx.x;
  if (i < N * 2) { segmax[i] = -1e30f; segsum[i] = 0.f; }
  if (i < N * 100) outacc[i] = 0.f;
}

// ---- node projections: x = memory[n_id]; q/k/v = x@W + b; z0 = x@Ws + bs --
__global__ void __launch_bounds__(128)
tgn_node_proj(const float* __restrict__ mem, const int* __restrict__ n_id, int N,
              const _Float16* __restrict__ WqPk, const float* __restrict__ bq, float* __restrict__ qout,
              const _Float16* __restrict__ WkPk, const float* __restrict__ bk, float* __restrict__ kout,
              const _Float16* __restrict__ WvPk, const float* __restrict__ bv, float* __restrict__ vout,
              const _Float16* __restrict__ WsPk, const float* __restrict__ bs, float* __restrict__ sout) {
  __shared__ _Float16 sx[4][16][128];
  const int w = threadIdx.x >> 5, lane = threadIdx.x & 31;
  const int tile = blockIdx.x * 4 + w;
  _Float16* at = &sx[w][0][0];
  for (int idx = lane; idx < 16 * 128; idx += 32) {
    int r = idx >> 7, c = idx & 127;
    int row = tile * 16 + r;
    float v = 0.f;
    if (row < N && c < 100) v = mem[(size_t)n_id[row] * 100 + c];
    at[idx] = (_Float16)v;
  }
  __syncthreads();
  v16h aF[4];
#pragma unroll
  for (int kt = 0; kt < 4; ++kt) aF[kt] = wmma_load_a(at, lane, kt, 128);
  const int n = lane & 15;
  const int mbase = (lane & 16) ? 8 : 0;
  const bool fullTile = (tile * 16 + 16 <= N);   // wave-uniform
#pragma unroll 1
  for (int nt = 0; nt < 7; ++nt) {
    const int col = nt * 16 + n;
    const bool cok = col < 100;
    auto do_w = [&](const _Float16* Wpk, const float* bias, float* out) {
      // preload the whole B chain, then run the WMMA chain
      v16h b0 = wmma_load_b_pk(Wpk, lane, 0, nt, 4);
      v16h b1 = wmma_load_b_pk(Wpk, lane, 1, nt, 4);
      v16h b2 = wmma_load_b_pk(Wpk, lane, 2, nt, 4);
      v16h b3 = wmma_load_b_pk(Wpk, lane, 3, nt, 4);
      v8f acc;
      float bz = cok ? bias[col] : 0.f;
#pragma unroll
      for (int r = 0; r < 8; ++r) acc[r] = bz;
      acc = wmma_mac(aF[0], b0, acc);
      acc = wmma_mac(aF[1], b1, acc);
      acc = wmma_mac(aF[2], b2, acc);
      acc = wmma_mac(aF[3], b3, acc);
      if (fullTile) {
        if (cok) {
          float* p = out + (size_t)(tile * 16 + mbase) * 100 + col;
#pragma unroll
          for (int r = 0; r < 8; ++r) p[(size_t)r * 100] = acc[r];
        }
      } else {
#pragma unroll
        for (int r = 0; r < 8; ++r) {
          int row = tile * 16 + mbase + r;
          if (row < N && cok) out[(size_t)row * 100 + col] = acc[r];
        }
      }
    };
    do_w(WqPk, bq, qout);
    do_w(WkPk, bk, kout);
    do_w(WvPk, bv, vout);
    do_w(WsPk, bs, sout);
  }
}

// ---- edge features (cos time-enc || msg) and e = edge_attr @ We -----------
__global__ void __launch_bounds__(128)
tgn_edge_gemm(const int* __restrict__ n_id, const int* __restrict__ src_e,
              const int* __restrict__ e_id, const float* __restrict__ t_store,
              const float* __restrict__ msg_store, const float* __restrict__ last_upd,
              const float* __restrict__ wt, const float* __restrict__ bt,
              const _Float16* __restrict__ WePk, float* __restrict__ ebuf, int E) {
  __shared__ _Float16 se[4][16][256];
  __shared__ float srel[4][16];
  const int w = threadIdx.x >> 5, lane = threadIdx.x & 31;
  const int tile = blockIdx.x * 4 + w;
  if (lane < 16) {
    int edge = tile * 16 + lane;
    float rel = 0.f;
    if (edge < E) rel = last_upd[n_id[src_e[edge]]] - t_store[e_id[edge]];
    srel[w][lane] = rel;
  }
  __syncthreads();
  _Float16* at = &se[w][0][0];
  for (int idx = lane; idx < 16 * 256; idx += 32) {
    int r = idx >> 8, c = idx & 255;
    int edge = tile * 16 + r;
    float v = 0.f;
    if (edge < E) {
      if (c < 100)      v = __cosf(srel[w][r] * wt[c] + bt[c]);
      else if (c < 200) v = msg_store[(size_t)e_id[edge] * 100 + (c - 100)];
    }
    at[idx] = (_Float16)v;
  }
  __syncthreads();
  v16h aF[8];
#pragma unroll
  for (int kt = 0; kt < 8; ++kt) aF[kt] = wmma_load_a(at, lane, kt, 256);
  const int n = lane & 15;
  const int mbase = (lane & 16) ? 8 : 0;
  const bool fullTile = (tile * 16 + 16 <= E);   // wave-uniform
#pragma unroll 1
  for (int nt = 0; nt < 7; ++nt) {
    const int col = nt * 16 + n;
    const bool cok = col < 100;
    v16h bF[8];
#pragma unroll
    for (int kt = 0; kt < 8; ++kt) bF[kt] = wmma_load_b_pk(WePk, lane, kt, nt, 8);
    v8f acc;
#pragma unroll
    for (int r = 0; r < 8; ++r) acc[r] = 0.f;
#pragma unroll
    for (int kt = 0; kt < 8; ++kt) acc = wmma_mac(aF[kt], bF[kt], acc);
    if (fullTile) {
      if (cok) {
        float* p = ebuf + (size_t)(tile * 16 + mbase) * 100 + col;
#pragma unroll
        for (int r = 0; r < 8; ++r) p[(size_t)r * 100] = acc[r];
      }
    } else {
#pragma unroll
      for (int r = 0; r < 8; ++r) {
        int edge = tile * 16 + mbase + r;
        if (edge < E && cok) ebuf[(size_t)edge * 100 + col] = acc[r];
      }
    }
  }
}

// ---- attention: logits + segment max ---------------------------------------
__global__ void tgn_logits(const float* __restrict__ qbuf, const float* __restrict__ kbuf,
                           const float* __restrict__ ebuf, const int* __restrict__ src_e,
                           const int* __restrict__ dst_e, float* __restrict__ logits,
                           float* __restrict__ segmax, int E) {
  int t = blockIdx.x * blockDim.x + threadIdx.x;
  if (t >= E * 2) return;
  int edge = t >> 1, h = t & 1;
  int s = src_e[edge], d = dst_e[edge];
  const float* q  = qbuf + (size_t)d * 100 + h * 50;
  const float* kk = kbuf + (size_t)s * 100 + h * 50;
  const float* ee = ebuf + (size_t)edge * 100 + h * 50;
  float acc = 0.f;
#pragma unroll 5
  for (int i = 0; i < 50; ++i) acc += q[i] * (kk[i] + ee[i]);
  float logit = acc * 0.1414213562373095f;  // 1/sqrt(50)
  logits[t] = logit;
  atomicMaxF(&segmax[(size_t)d * 2 + h], logit);
}

// ---- attention: exp + segment sum ------------------------------------------
__global__ void tgn_exps(const float* __restrict__ logits, const float* __restrict__ segmax,
                         const int* __restrict__ dst_e, float* __restrict__ exbuf,
                         float* __restrict__ segsum, int E) {
  int t = blockIdx.x * blockDim.x + threadIdx.x;
  if (t >= E * 2) return;
  int edge = t >> 1, h = t & 1;
  int d = dst_e[edge];
  float ex = __expf(logits[t] - segmax[(size_t)d * 2 + h]);
  exbuf[t] = ex;
  atomicAdd(&segsum[(size_t)d * 2 + h], ex);
}

// ---- attention: weighted scatter-add ---------------------------------------
__global__ void tgn_scatter(const float* __restrict__ exbuf, const float* __restrict__ segsum,
                            const float* __restrict__ vbuf, const float* __restrict__ ebuf,
                            const int* __restrict__ src_e, const int* __restrict__ dst_e,
                            float* __restrict__ outacc, int E) {
  long long t = (long long)blockIdx.x * blockDim.x + threadIdx.x;
  if (t >= (long long)E * 100) return;
  int edge = (int)(t / 100), c = (int)(t % 100);
  int h = (c >= 50);
  int s = src_e[edge], d = dst_e[edge];
  float alpha = exbuf[(size_t)edge * 2 + h] / (segsum[(size_t)d * 2 + h] + 1e-16f);
  float val = (vbuf[(size_t)s * 100 + c] + ebuf[(size_t)edge * 100 + c]) * alpha;
  atomicAdd(&outacc[(size_t)d * 100 + c], val);
}

__global__ void tgn_zadd(float* __restrict__ z, const float* __restrict__ outacc, int total) {
  int i = blockIdx.x * blockDim.x + threadIdx.x;
  if (i < total) z[i] += outacc[i];
}

// ---- link predictor: relu(zs@Wls+bls + zx@Wld+bld) @ Wlf + blf -------------
__global__ void __launch_bounds__(128)
tgn_linkpred(const float* __restrict__ z, const int* __restrict__ srcI,
             const int* __restrict__ dstI, const int* __restrict__ negI,
             const _Float16* __restrict__ WlsPk, const float* __restrict__ bls,
             const _Float16* __restrict__ WldPk, const float* __restrict__ bld,
             const float* __restrict__ Wlf, const float* __restrict__ blf,
             float* __restrict__ outp, int B) {
  __shared__ _Float16 zls[4][3][16][128];
  const int w = threadIdx.x >> 5, lane = threadIdx.x & 31;
  const int tile = blockIdx.x * 4 + w;
  for (int idx = lane; idx < 3 * 16 * 128; idx += 32) {
    int which = idx / (16 * 128);
    int rem = idx - which * (16 * 128);
    int r = rem >> 7, c = rem & 127;
    int row = tile * 16 + r;
    const int* sel = (which == 0) ? srcI : (which == 1) ? dstI : negI;
    float v = 0.f;
    if (row < B && c < 100) v = z[(size_t)sel[row] * 100 + c];
    zls[w][which][r][c] = (_Float16)v;
  }
  __syncthreads();
  v16h aS[4], aD[4], aN[4];
#pragma unroll
  for (int kt = 0; kt < 4; ++kt) {
    aS[kt] = wmma_load_a(&zls[w][0][0][0], lane, kt, 128);
    aD[kt] = wmma_load_a(&zls[w][1][0][0], lane, kt, 128);
    aN[kt] = wmma_load_a(&zls[w][2][0][0], lane, kt, 128);
  }
  const int n = lane & 15;
  const int mbase = (lane & 16) ? 8 : 0;
  float ppos[8], pneg[8];
#pragma unroll
  for (int r = 0; r < 8; ++r) { ppos[r] = 0.f; pneg[r] = 0.f; }
#pragma unroll 1
  for (int nt = 0; nt < 7; ++nt) {
    const int col = nt * 16 + n;
    const bool cok = col < 100;
    float bsv = cok ? bls[col] : 0.f;
    float bdv = cok ? bld[col] : 0.f;
    // preload both B chains (Wld fragments shared by dst and neg paths)
    v16h bS[4], bD[4];
#pragma unroll
    for (int kt = 0; kt < 4; ++kt) {
      bS[kt] = wmma_load_b_pk(WlsPk, lane, kt, nt, 4);
      bD[kt] = wmma_load_b_pk(WldPk, lane, kt, nt, 4);
    }
    v8f accs, accd, accn;
#pragma unroll
    for (int r = 0; r < 8; ++r) { accs[r] = bsv; accd[r] = bdv; accn[r] = bdv; }
#pragma unroll
    for (int kt = 0; kt < 4; ++kt) {
      accs = wmma_mac(aS[kt], bS[kt], accs);
      accd = wmma_mac(aD[kt], bD[kt], accd);
      accn = wmma_mac(aN[kt], bD[kt], accn);
    }
    float wv = cok ? Wlf[col] : 0.f;
#pragma unroll
    for (int r = 0; r < 8; ++r) {
      ppos[r] += fmaxf(accs[r] + accd[r], 0.f) * wv;
      pneg[r] += fmaxf(accs[r] + accn[r], 0.f) * wv;
    }
  }
  // reduce 16 columns across each half-wave (lanes 0-15 / 16-31 stay separate)
#pragma unroll
  for (int r = 0; r < 8; ++r) {
#pragma unroll
    for (int off = 8; off >= 1; off >>= 1) {
      ppos[r] += __shfl_xor(ppos[r], off, 32);
      pneg[r] += __shfl_xor(pneg[r], off, 32);
    }
  }
  float b0 = blf[0];
  if ((lane & 15) == 0) {
#pragma unroll
    for (int r = 0; r < 8; ++r) {
      int row = tile * 16 + mbase + r;
      if (row < B) {
        outp[row]     = ppos[r] + b0;
        outp[B + row] = pneg[r] + b0;
      }
    }
  }
}

// ---------------------------------------------------------------------------
extern "C" void kernel_launch(void* const* d_in, const int* in_sizes, int n_in,
                              void* d_out, int out_size, void* d_ws, size_t ws_size,
                              hipStream_t stream) {
  const int*   src       = (const int*)d_in[0];
  const int*   dst       = (const int*)d_in[1];
  const int*   neg       = (const int*)d_in[2];
  const int*   n_id      = (const int*)d_in[3];
  const int*   eidx      = (const int*)d_in[4];
  const int*   e_id      = (const int*)d_in[5];
  const float* msg_store = (const float*)d_in[6];
  const float* t_store   = (const float*)d_in[7];
  const float* memf      = (const float*)d_in[8];
  const float* last_upd  = (const float*)d_in[9];
  const float* wt  = (const float*)d_in[10];
  const float* bt  = (const float*)d_in[11];
  const float* Wq  = (const float*)d_in[12]; const float* bq  = (const float*)d_in[13];
  const float* Wk  = (const float*)d_in[14]; const float* bk  = (const float*)d_in[15];
  const float* Wv  = (const float*)d_in[16]; const float* bv  = (const float*)d_in[17];
  const float* We  = (const float*)d_in[18];
  const float* Ws  = (const float*)d_in[19]; const float* bs  = (const float*)d_in[20];
  const float* Wls = (const float*)d_in[21]; const float* bls = (const float*)d_in[22];
  const float* Wld = (const float*)d_in[23]; const float* bld = (const float*)d_in[24];
  const float* Wlf = (const float*)d_in[25]; const float* blf = (const float*)d_in[26];

  const int B = in_sizes[0];
  const int N = in_sizes[3];
  const int E = in_sizes[5];
  const int* src_e = eidx;
  const int* dst_e = eidx + E;

  // workspace carve-out
  char* ws = (char*)d_ws;
  size_t off = 0;
  auto alloc_f = [&](size_t nelem) -> float* {
    float* p = (float*)(ws + off);
    off += nelem * sizeof(float);
    off = (off + 255) & ~(size_t)255;
    return p;
  };
  auto alloc_h = [&](size_t nelem) -> _Float16* {
    _Float16* p = (_Float16*)(ws + off);
    off += nelem * sizeof(_Float16);
    off = (off + 255) & ~(size_t)255;
    return p;
  };
  float* qbuf   = alloc_f((size_t)N * 100);
  float* kbuf   = alloc_f((size_t)N * 100);
  float* vbuf   = alloc_f((size_t)N * 100);
  float* zbuf   = alloc_f((size_t)N * 100);   // skip connection, later z
  float* outacc = alloc_f((size_t)N * 100);
  float* ebuf   = alloc_f((size_t)E * 100);
  float* logitsb= alloc_f((size_t)E * 2);
  float* exb    = alloc_f((size_t)E * 2);
  float* segmax = alloc_f((size_t)N * 2);
  float* segsum = alloc_f((size_t)N * 2);
  const int packTot  = 7 * 4 * 32 * 16;   // 14336 halves, nkt=4 weights
  const int packTotE = 7 * 8 * 32 * 16;   // 28672 halves, nkt=8 (We)
  _Float16* WqPk  = alloc_h(packTot);
  _Float16* WkPk  = alloc_h(packTot);
  _Float16* WvPk  = alloc_h(packTot);
  _Float16* WsPk  = alloc_h(packTot);
  _Float16* WlsPk = alloc_h(packTot);
  _Float16* WldPk = alloc_h(packTot);
  _Float16* WePk  = alloc_h(packTotE);
  (void)ws_size; (void)n_in; (void)out_size;

  const int TPB = 256;
  tgn_pack_w_frag<<<(packTot + TPB - 1) / TPB, TPB, 0, stream>>>(Wq,  WqPk,  100, 100, 4, packTot);
  tgn_pack_w_frag<<<(packTot + TPB - 1) / TPB, TPB, 0, stream>>>(Wk,  WkPk,  100, 100, 4, packTot);
  tgn_pack_w_frag<<<(packTot + TPB - 1) / TPB, TPB, 0, stream>>>(Wv,  WvPk,  100, 100, 4, packTot);
  tgn_pack_w_frag<<<(packTot + TPB - 1) / TPB, TPB, 0, stream>>>(Ws,  WsPk,  100, 100, 4, packTot);
  tgn_pack_w_frag<<<(packTot + TPB - 1) / TPB, TPB, 0, stream>>>(Wls, WlsPk, 100, 100, 4, packTot);
  tgn_pack_w_frag<<<(packTot + TPB - 1) / TPB, TPB, 0, stream>>>(Wld, WldPk, 100, 100, 4, packTot);
  tgn_pack_w_frag<<<(packTotE + TPB - 1) / TPB, TPB, 0, stream>>>(We, WePk, 200, 100, 8, packTotE);

  tgn_init<<<((unsigned)(N * 100) + TPB - 1) / TPB, TPB, 0, stream>>>(segmax, segsum, outacc, N);

  const int ntile_n = (N + 15) / 16;
  tgn_node_proj<<<(ntile_n + 3) / 4, 128, 0, stream>>>(
      memf, n_id, N, WqPk, bq, qbuf, WkPk, bk, kbuf, WvPk, bv, vbuf, WsPk, bs, zbuf);

  const int ntile_e = (E + 15) / 16;
  tgn_edge_gemm<<<(ntile_e + 3) / 4, 128, 0, stream>>>(
      n_id, src_e, e_id, t_store, msg_store, last_upd, wt, bt, WePk, ebuf, E);

  tgn_logits<<<((unsigned)(E * 2) + TPB - 1) / TPB, TPB, 0, stream>>>(
      qbuf, kbuf, ebuf, src_e, dst_e, logitsb, segmax, E);
  tgn_exps<<<((unsigned)(E * 2) + TPB - 1) / TPB, TPB, 0, stream>>>(
      logitsb, segmax, dst_e, exb, segsum, E);

  long long scatTot = (long long)E * 100;
  tgn_scatter<<<(unsigned)((scatTot + TPB - 1) / TPB), TPB, 0, stream>>>(
      exb, segsum, vbuf, ebuf, src_e, dst_e, outacc, E);

  tgn_zadd<<<((unsigned)(N * 100) + TPB - 1) / TPB, TPB, 0, stream>>>(zbuf, outacc, N * 100);

  const int ntile_b = (B + 15) / 16;
  tgn_linkpred<<<(ntile_b + 3) / 4, 128, 0, stream>>>(
      zbuf, src, dst, neg, WlsPk, bls, WldPk, bld, Wlf, blf, (float*)d_out, B);
}
